// Attention_68401649156671
// MI455X (gfx1250) — compile-verified, matrix operand order checked by reading, not run on hardware
//
#include <hip/hip_runtime.h>

typedef __attribute__((ext_vector_type(16))) __bf16        v16bf;
typedef __attribute__((ext_vector_type(8)))  float         v8f;
typedef __attribute__((ext_vector_type(4)))  unsigned int  v4u;
typedef unsigned short u16;
typedef unsigned int   u32;
typedef unsigned long long u64;

#define DIM    1024
#define NHEADS 16
#define HDIM   64
#define SEQ    2048
#define BATCH  2
#define MTOT   (BATCH * SEQ)   /* 4096 token rows */
#define ATT_SCALE 0.125f       /* 64^-0.5 */

// ---- helpers ---------------------------------------------------------------

__device__ __forceinline__ u16 f2bf(float f) {
  u32 u = __builtin_bit_cast(u32, f);
  return (u16)((u + 0x7FFFu + ((u >> 16) & 1u)) >> 16);  // round-to-nearest-even
}

union Frag {
  v16bf v;
  v4u   q[2];
  u16   u[16];
};

// Load a 16x32 bf16 A/B fragment (ISA 16-bit layout) from row-major memory:
// lane L (<16): row=row0+L,    K = k0+{0..7, 16..23}
// lane L (>=16): row=row0+L-16, K = k0+{8..15, 24..31}
__device__ __forceinline__ v16bf load_frag(const u16* base, int row0, int ld, int k0) {
  const int lane = threadIdx.x & 31;
  const u16* p = base + (size_t)(row0 + (lane & 15)) * ld + k0 + ((lane >> 4) << 3);
  Frag f;
  f.q[0] = *(const v4u*)(p);
  f.q[1] = *(const v4u*)(p + 16);
  return f.v;
}

__device__ __forceinline__ v8f wmma_bf16(v16bf a, v16bf b, v8f c) {
  return __builtin_amdgcn_wmma_f32_16x16x32_bf16(
      /*neg_a=*/false, a, /*neg_b=*/false, b,
      /*c_mod=*/(short)0, c, /*reuse_a=*/false, /*reuse_b=*/false);
}

// Asynchronously copy a contiguous 4KB block (global -> LDS) with the CDNA5
// async DMA path: 8x global_load_async_to_lds_b128, each lane moving 16B.
// The instruction offset applies to BOTH the LDS dest and global source
// (CDNA5 ISA), so one address pair covers the tile.
__device__ __forceinline__ void async_copy_4k(const u16* gsrc, u16* ldst, int lane) {
  u32 lbase = (u32)(uintptr_t)ldst + (u32)lane * 16;   // flat[31:0] == LDS addr
  u64 gbase = (u64)(uintptr_t)gsrc + (u64)lane * 16;
  asm volatile("global_load_async_to_lds_b128 %0, %1, off"              :: "v"(lbase), "v"(gbase) : "memory");
  asm volatile("global_load_async_to_lds_b128 %0, %1, off offset:512"   :: "v"(lbase), "v"(gbase) : "memory");
  asm volatile("global_load_async_to_lds_b128 %0, %1, off offset:1024"  :: "v"(lbase), "v"(gbase) : "memory");
  asm volatile("global_load_async_to_lds_b128 %0, %1, off offset:1536"  :: "v"(lbase), "v"(gbase) : "memory");
  asm volatile("global_load_async_to_lds_b128 %0, %1, off offset:2048"  :: "v"(lbase), "v"(gbase) : "memory");
  asm volatile("global_load_async_to_lds_b128 %0, %1, off offset:2560"  :: "v"(lbase), "v"(gbase) : "memory");
  asm volatile("global_load_async_to_lds_b128 %0, %1, off offset:3072"  :: "v"(lbase), "v"(gbase) : "memory");
  asm volatile("global_load_async_to_lds_b128 %0, %1, off offset:3584"  :: "v"(lbase), "v"(gbase) : "memory");
}

// ---- precision-conversion kernels ------------------------------------------

__global__ __launch_bounds__(256) void cvt_bf16(const float* __restrict__ in,
                                                u16* __restrict__ out, int n) {
  int i = blockIdx.x * 256 + threadIdx.x;
  if (i < n) out[i] = f2bf(in[i]);
}

// wt[n*K + k] = bf16(w[k*N + n])  (transpose to N-major for B-fragment loads)
__global__ __launch_bounds__(256) void cvt_transpose(const float* __restrict__ w,
                                                     u16* __restrict__ wt,
                                                     int K, int N) {
  int i = blockIdx.x * 256 + threadIdx.x;
  if (i < K * N) {
    int n = i / K;
    int k = i - n * K;
    wt[i] = f2bf(w[(size_t)k * N + n]);
  }
}

// ---- QKV projection GEMM: [4096,1024] @ [1024,3072] + bias -----------------
// wave tile: 32(M) x 64(N); k-loop unrolled 2x with even/odd ping-pong
// fragment sets (no register rotation movs); epilogue adds bias and
// scatter-stores bf16 into Q, K, V^T layouts.

__global__ __launch_bounds__(256) void qkv_gemm(const u16* __restrict__ xbf,
                                                const u16* __restrict__ wt,
                                                const float* __restrict__ bias,
                                                u16* __restrict__ Qb,
                                                u16* __restrict__ Kb,
                                                u16* __restrict__ Vtb) {
  const int lane  = threadIdx.x & 31;
  const int wglob = blockIdx.x * 8 + (threadIdx.x >> 5);
  const int NT    = (3 * DIM) / 64;        // 48 N-tiles of width 64
  const int mt    = wglob / NT;            // 0..127 (32-row tiles)
  const int nt    = wglob % NT;
  const int m0    = mt * 32;
  const int n0    = nt * 64;

  v8f acc[2][4] = {};
  v16bf aE[2], bE[4], aO[2], bO[4];
#pragma unroll
  for (int i = 0; i < 2; ++i) aE[i] = load_frag(xbf, m0 + i * 16, DIM, 0);
#pragma unroll
  for (int j = 0; j < 4; ++j) bE[j] = load_frag(wt, n0 + j * 16, DIM, 0);

  for (int k0 = 0; k0 < DIM; k0 += 64) {
#pragma unroll
    for (int i = 0; i < 2; ++i) aO[i] = load_frag(xbf, m0 + i * 16, DIM, k0 + 32);
#pragma unroll
    for (int j = 0; j < 4; ++j) bO[j] = load_frag(wt, n0 + j * 16, DIM, k0 + 32);
#pragma unroll
    for (int i = 0; i < 2; ++i)
#pragma unroll
      for (int j = 0; j < 4; ++j)
        acc[i][j] = wmma_bf16(aE[i], bE[j], acc[i][j]);
    if (k0 + 64 < DIM) {
#pragma unroll
      for (int i = 0; i < 2; ++i) aE[i] = load_frag(xbf, m0 + i * 16, DIM, k0 + 64);
#pragma unroll
      for (int j = 0; j < 4; ++j) bE[j] = load_frag(wt, n0 + j * 16, DIM, k0 + 64);
    }
#pragma unroll
    for (int i = 0; i < 2; ++i)
#pragma unroll
      for (int j = 0; j < 4; ++j)
        acc[i][j] = wmma_bf16(aO[i], bO[j], acc[i][j]);
  }

  const int half = (lane >> 4) << 3;
#pragma unroll
  for (int i = 0; i < 2; ++i) {
#pragma unroll
    for (int j = 0; j < 4; ++j) {
#pragma unroll
      for (int r = 0; r < 8; ++r) {
        int row = m0 + i * 16 + r + half;           // token index in [0,4096)
        int col = n0 + j * 16 + (lane & 15);        // feature in [0,3072)
        u16 h = f2bf(acc[i][j][r] + bias[col]);
        int sel = col >> 10;            // 0=q 1=k 2=v
        int hh  = (col >> 6) & 15;      // head
        int dd  = col & 63;             // dim within head
        int b   = row >> 11;
        int s   = row & (SEQ - 1);
        size_t bh = (size_t)(b * NHEADS + hh);
        if (sel == 0)      Qb [(bh * SEQ + s) * HDIM + dd] = h;
        else if (sel == 1) Kb [(bh * SEQ + s) * HDIM + dd] = h;
        else               Vtb[(bh * HDIM + dd) * SEQ + s] = h;   // V transposed
      }
    }
  }
}

// ---- flash attention: one wave per (b,h,16-query tile) ---------------------
// K tiles (contiguous 4KB) are double-buffered in LDS via async-DMA loads;
// V^T fragments load straight from global (strided tile); P goes through an
// LDS transpose (C-layout -> A-layout).

__global__ __launch_bounds__(128) void flash_attn(const u16* __restrict__ Qb,
                                                  const u16* __restrict__ Kb,
                                                  const u16* __restrict__ Vtb,
                                                  u16* __restrict__ attn) {
  __shared__ __align__(16) u16 kstage[4][2][32 * HDIM];  // 4 waves x 2 x 4KB
  __shared__ __align__(16) u16 pstage[4][16 * 32];       // P transpose tiles
  const int lane = threadIdx.x & 31;
  const int wave = threadIdx.x >> 5;
  const int wt   = blockIdx.x * 4 + wave;         // 0..4095
  const int qt   = wt & (SEQ / 16 - 1);           // query tile 0..127
  const int bh   = wt >> 7;                       // 0..31

  const u16* Qp = Qb  + (size_t)bh * SEQ * HDIM;
  const u16* Kp = Kb  + (size_t)bh * SEQ * HDIM;
  const u16* Vp = Vtb + (size_t)bh * HDIM * SEQ;
  u16* lp  = pstage[wave];
  u16* kb0 = kstage[wave][0];
  u16* kb1 = kstage[wave][1];

  const v16bf qf0 = load_frag(Qp, qt * 16, HDIM, 0);
  const v16bf qf1 = load_frag(Qp, qt * 16, HDIM, 32);

  float m[8], l[8];
  v8f o[4] = {};
#pragma unroll
  for (int r = 0; r < 8; ++r) { m[r] = -1e30f; l[r] = 0.f; }

  // preload first K tile into buffer 0
  async_copy_4k(Kp, kb0, lane);

  for (int k0 = 0; k0 < SEQ; k0 += 32) {
    const int t = k0 >> 5;
    u16* kc = (t & 1) ? kb1 : kb0;   // tile being consumed
    u16* kn = (t & 1) ? kb0 : kb1;   // tile being filled

    if (k0 + 32 < SEQ) {
      // start DMA of the next K tile, then wait only for the current one
      // (async ops complete in order: <=8 outstanding => current tile landed)
      async_copy_4k(Kp + (size_t)(k0 + 32) * HDIM, kn, lane);
      asm volatile("s_wait_asynccnt 0x8" ::: "memory");
      // prefetch next V^T key-chunk into cache
      __builtin_prefetch(Vp + (size_t)lane * SEQ + (k0 + 32), 0, 3);
      __builtin_prefetch(Vp + (size_t)(lane + 32) * SEQ + (k0 + 32), 0, 3);
    } else {
      asm volatile("s_wait_asynccnt 0x0" ::: "memory");
    }

    // scores for 32 keys: S = Q K^T  (fragments from the LDS K tile)
    v8f s0 = {}, s1 = {};
    s0 = wmma_bf16(qf0, load_frag(kc, 0,  HDIM, 0),  s0);
    s0 = wmma_bf16(qf1, load_frag(kc, 0,  HDIM, 32), s0);
    s1 = wmma_bf16(qf0, load_frag(kc, 16, HDIM, 0),  s1);
    s1 = wmma_bf16(qf1, load_frag(kc, 16, HDIM, 32), s1);

    // online softmax per row (row r lives in VGPR r, N across 16-lane half)
#pragma unroll
    for (int r = 0; r < 8; ++r) {
      float a0 = s0[r] * ATT_SCALE;
      float a1 = s1[r] * ATT_SCALE;
      float v = fmaxf(a0, a1);
      v = fmaxf(v, __shfl_xor(v, 8, 32));
      v = fmaxf(v, __shfl_xor(v, 4, 32));
      v = fmaxf(v, __shfl_xor(v, 2, 32));
      v = fmaxf(v, __shfl_xor(v, 1, 32));
      float mnew  = fmaxf(m[r], v);
      float alpha = __expf(m[r] - mnew);
      float p0 = __expf(a0 - mnew);
      float p1 = __expf(a1 - mnew);
      float rs = p0 + p1;
      rs += __shfl_xor(rs, 8, 32);
      rs += __shfl_xor(rs, 4, 32);
      rs += __shfl_xor(rs, 2, 32);
      rs += __shfl_xor(rs, 1, 32);
      l[r] = l[r] * alpha + rs;
      m[r] = mnew;
#pragma unroll
      for (int j = 0; j < 4; ++j) o[j][r] *= alpha;
      // stage P (C-layout) into LDS for the layout transpose
      int row = r + ((lane >> 4) << 3);
      lp[row * 32 +      (lane & 15)] = f2bf(p0);
      lp[row * 32 + 16 + (lane & 15)] = f2bf(p1);
    }
    asm volatile("s_wait_dscnt 0x0" ::: "memory");   // same-wave DS RAW

    // reload P as a 16x32 A-fragment (ds_load_b128 x2)
    Frag pf;
    const u16* tp = lp + (lane & 15) * 32 + ((lane >> 4) << 3);
    pf.q[0] = *(const v4u*)(tp);
    pf.q[1] = *(const v4u*)(tp + 16);

    // O += P @ V  (B-fragments from V^T rows: contiguous in key index)
#pragma unroll
    for (int j = 0; j < 4; ++j) {
      v16bf vf = load_frag(Vp, j * 16, SEQ, k0);
      o[j] = wmma_bf16(pf.v, vf, o[j]);
    }
    // same-wave DS ops are in-order: next iteration's P stores cannot pass
    // this iteration's ds_load_b128, so no extra wait needed here.
  }

  // epilogue: normalize and store to attn[B,S,H*d] in bf16
  const int b = bh >> 4;
  const int h = bh & 15;
#pragma unroll
  for (int r = 0; r < 8; ++r) {
    float inv = 1.f / l[r];
    int s = qt * 16 + r + ((lane >> 4) << 3);
    size_t rowoff = ((size_t)(b * SEQ + s)) * DIM + h * HDIM;
#pragma unroll
    for (int j = 0; j < 4; ++j)
      attn[rowoff + j * 16 + (lane & 15)] = f2bf(o[j][r] * inv);
  }
}

// ---- output projection GEMM: [4096,1024] @ [1024,1024] + bias -> f32 -------
// same 32x64 ping-pong wave tile as qkv_gemm.

__global__ __launch_bounds__(256) void out_gemm(const u16* __restrict__ abf,
                                                const u16* __restrict__ wt,
                                                const float* __restrict__ bias,
                                                float* __restrict__ out) {
  const int lane  = threadIdx.x & 31;
  const int wglob = blockIdx.x * 8 + (threadIdx.x >> 5);
  const int NT    = DIM / 64;              // 16
  const int mt    = wglob / NT;            // 0..127
  const int nt    = wglob % NT;
  const int m0    = mt * 32;
  const int n0    = nt * 64;

  v8f acc[2][4] = {};
  v16bf aE[2], bE[4], aO[2], bO[4];
#pragma unroll
  for (int i = 0; i < 2; ++i) aE[i] = load_frag(abf, m0 + i * 16, DIM, 0);
#pragma unroll
  for (int j = 0; j < 4; ++j) bE[j] = load_frag(wt, n0 + j * 16, DIM, 0);

  for (int k0 = 0; k0 < DIM; k0 += 64) {
#pragma unroll
    for (int i = 0; i < 2; ++i) aO[i] = load_frag(abf, m0 + i * 16, DIM, k0 + 32);
#pragma unroll
    for (int j = 0; j < 4; ++j) bO[j] = load_frag(wt, n0 + j * 16, DIM, k0 + 32);
#pragma unroll
    for (int i = 0; i < 2; ++i)
#pragma unroll
      for (int j = 0; j < 4; ++j)
        acc[i][j] = wmma_bf16(aE[i], bE[j], acc[i][j]);
    if (k0 + 64 < DIM) {
#pragma unroll
      for (int i = 0; i < 2; ++i) aE[i] = load_frag(abf, m0 + i * 16, DIM, k0 + 64);
#pragma unroll
      for (int j = 0; j < 4; ++j) bE[j] = load_frag(wt, n0 + j * 16, DIM, k0 + 64);
    }
#pragma unroll
    for (int i = 0; i < 2; ++i)
#pragma unroll
      for (int j = 0; j < 4; ++j)
        acc[i][j] = wmma_bf16(aO[i], bO[j], acc[i][j]);
  }

  const int half = (lane >> 4) << 3;
#pragma unroll
  for (int i = 0; i < 2; ++i) {
#pragma unroll
    for (int j = 0; j < 4; ++j) {
#pragma unroll
      for (int r = 0; r < 8; ++r) {
        int row = m0 + i * 16 + r + half;
        int col = n0 + j * 16 + (lane & 15);
        out[(size_t)row * DIM + col] = acc[i][j][r] + bias[col];
      }
    }
  }
}

// ---- host launch -----------------------------------------------------------

extern "C" void kernel_launch(void* const* d_in, const int* in_sizes, int n_in,
                              void* d_out, int out_size, void* d_ws, size_t ws_size,
                              hipStream_t stream) {
  const float* x     = (const float*)d_in[0];
  const float* w_qkv = (const float*)d_in[1];
  const float* b_qkv = (const float*)d_in[2];
  const float* w_out = (const float*)d_in[3];
  const float* b_out = (const float*)d_in[4];
  float* out = (float*)d_out;

  char* ws = (char*)d_ws;
  size_t off = 0;
  u16* xbf   = (u16*)(ws + off); off += (size_t)MTOT * DIM * 2;                       // 8 MB
  u16* wqkvt = (u16*)(ws + off); off += (size_t)3 * DIM * DIM * 2;                    // 6 MB
  u16* woutt = (u16*)(ws + off); off += (size_t)DIM * DIM * 2;                        // 2 MB
  u16* Qb    = (u16*)(ws + off); off += (size_t)BATCH * NHEADS * SEQ * HDIM * 2;      // 8 MB
  u16* Kb    = (u16*)(ws + off); off += (size_t)BATCH * NHEADS * SEQ * HDIM * 2;      // 8 MB
  u16* Vtb   = (u16*)(ws + off); off += (size_t)BATCH * NHEADS * SEQ * HDIM * 2;      // 8 MB
  u16* attn  = (u16*)(ws + off); off += (size_t)MTOT * DIM * 2;                       // 8 MB

  cvt_bf16<<<(MTOT * DIM) / 256, 256, 0, stream>>>(x, xbf, MTOT * DIM);
  cvt_transpose<<<(3 * DIM * DIM) / 256, 256, 0, stream>>>(w_qkv, wqkvt, DIM, 3 * DIM);
  cvt_transpose<<<(DIM * DIM) / 256, 256, 0, stream>>>(w_out, woutt, DIM, DIM);

  // (4096/32 M-tiles) * (3072/64 N-tiles) waves / 8 waves per block = 768 blocks
  qkv_gemm<<<(MTOT / 32) * ((3 * DIM) / 64) / 8, 256, 0, stream>>>(
      xbf, wqkvt, b_qkv, Qb, Kb, Vtb);

  // B*H*(S/16) = 4096 wave-tiles, 4 waves per block
  flash_attn<<<(BATCH * NHEADS * (SEQ / 16)) / 4, 128, 0, stream>>>(Qb, Kb, Vtb, attn);

  // (4096/32) * (1024/64) / 8 = 256 blocks
  out_gemm<<<(MTOT / 32) * (DIM / 64) / 8, 256, 0, stream>>>(attn, woutt, b_out, out);
}
